// AdaptiveBlock_81570018886345
// MI455X (gfx1250) — matrix operand order repaired and checked by reading, not compiled, not run on hardware
//
#include <hip/hip_runtime.h>
#include <hip/hip_bf16.h>
#include <math.h>

// ---------------------------------------------------------------------------
// Problem constants
// ---------------------------------------------------------------------------
#define BB    8
#define TT    8
#define BT    64          // B*T
#define HH    512
#define E2    512
#define NPIX  49
#define NPAD  64          // padded NPIX
#define CDIM  2048
#define VOCAB 10000

typedef __attribute__((ext_vector_type(16))) __bf16        bf16x16;
typedef __attribute__((ext_vector_type(8)))  float         f32x8;
typedef __attribute__((ext_vector_type(2)))  float         f32x2;
typedef __attribute__((ext_vector_type(4)))  unsigned int  u32x4;

// ---------------------------------------------------------------------------
// bf16 helpers
// ---------------------------------------------------------------------------
__device__ __forceinline__ unsigned int pack2(float a, float b) {
  union { unsigned int u; __bf16 h[2]; } t;
  t.h[0] = (__bf16)a;
  t.h[1] = (__bf16)b;
  return t.u;
}

union Frag { bf16x16 v; u32x4 q[2]; };

// 8 consecutive u32 (16 bf16) from LDS: two b128 loads
__device__ __forceinline__ bf16x16 ld_frag_b(const unsigned int* p) {
  Frag f;
  f.q[0] = *(const u32x4*)(p);
  f.q[1] = *(const u32x4*)(p + 4);
  return f.v;
}
// A fragment: groups at +0..3 and +8..11 relative to base
__device__ __forceinline__ bf16x16 ld_frag_a(const unsigned int* p) {
  Frag f;
  f.q[0] = *(const u32x4*)(p);
  f.q[1] = *(const u32x4*)(p + 8);
  return f.v;
}

// ---------------------------------------------------------------------------
// bf16-WMMA GEMM:  C[M,N] (fp32) = A[M,K] (fp32) * B[K,N] (fp32)
// Requirements: M multiple of 64 (grid.y = M/64), K multiple of 32,
//               N multiple of 16 (ragged last block allowed: tile-uniform).
// Block = 128 threads = 4 waves; block tile 64x64; LDS-staged packed bf16.
//   LDS A: [64 m][16 kpair]  (k contiguous pairs)
//   LDS B: [64 n][16 kpair]  (transposed so fragments are contiguous)
// Wave32 fragment maps used here (C/D as in CDNA5 ISA 7.12.2):
//   A: e=0..15 -> K=e+16*half
//   B: e=0..15 -> K=e+16*half (same as A; not the ISA table order)
//   C/D: acc[r] = D[r+8*half][lane%16]
// ---------------------------------------------------------------------------
__global__ void gemm_wmma_bf16(const float* __restrict__ A,
                               const float* __restrict__ Bm,
                               float* __restrict__ C,
                               int N, int K,
                               int lda, int ldb, int ldc,
                               int accumulate,
                               const float* __restrict__ bias) {
  __shared__ unsigned int sA[64 * 16];
  __shared__ unsigned int sB[64 * 16];

  const int tid    = threadIdx.x;
  const int lane   = tid & 31;
  const int wave   = tid >> 5;
  const int half   = lane >> 4;
  const int l16    = lane & 15;
  const int m_base = blockIdx.y * 64;
  const int n0     = blockIdx.x * 64;

  f32x8 acc[4] = {f32x8(0.f), f32x8(0.f), f32x8(0.f), f32x8(0.f)};

  const unsigned int* fragA = &sA[(wave * 16 + l16) * 16 + 8 * half];
  const unsigned int* fragB = &sB[l16 * 16 + 8 * half];

  for (int k0 = 0; k0 < K; k0 += 32) {
    __syncthreads();
    // ---- stage A: 64 rows x 32 k, packed pairs along k (coalesced b64) ----
#pragma unroll
    for (int it = 0; it < 8; ++it) {
      int idx = tid + it * 128;            // 0..1023
      int r = idx >> 4, p = idx & 15;
      const float* src = A + (size_t)(m_base + r) * lda + (k0 + 2 * p);
      f32x2 f = *(const f32x2*)src;
      sA[idx] = pack2(f.x, f.y);
    }
    // ---- stage B transposed: [n][kpair]; branchless N clamp ----
#pragma unroll
    for (int it = 0; it < 8; ++it) {
      int idx = tid + it * 128;
      int n = idx >> 4, p = idx & 15;
      int ng = n0 + n;
      int nc = (ng < N) ? ng : (N - 1);
      const float* bp = Bm + (size_t)(k0 + 2 * p) * ldb + nc;
      float v0 = bp[0];
      float v1 = bp[ldb];
      if (ng >= N) { v0 = 0.f; v1 = 0.f; }
      sB[idx] = pack2(v0, v1);
    }
    __syncthreads();

    bf16x16 a = ld_frag_b(fragA);
#pragma unroll
    for (int t = 0; t < 4; ++t) {
      bf16x16 b = ld_frag_b(fragB + t * 256);   // (16*t + l16)*16 + 8*half
      acc[t] = __builtin_amdgcn_wmma_f32_16x16x32_bf16(
          false, a, false, b, (short)0, acc[t], false, false);
    }
  }

  // ---- epilogue: tile-uniform N guard (N % 16 == 0) ----
#pragma unroll
  for (int t = 0; t < 4; ++t) {
    int nt = n0 + 16 * t;
    if (nt < N) {
      int n = nt + l16;
      float bv = bias ? bias[n] : 0.f;
      float* cp = C + (size_t)(m_base + wave * 16 + 8 * half) * ldc + n;
#pragma unroll
      for (int r = 0; r < 8; ++r) {
        float v = acc[t][r] + bv;
        if (accumulate) v += cp[(size_t)r * ldc];
        cp[(size_t)r * ldc] = v;
      }
    }
  }
}

// ---------------------------------------------------------------------------
// Padding / fill helpers
// ---------------------------------------------------------------------------
// dst[r, 0:dcols) = src[r, 0:scols) padded with zeros
__global__ void k_padcols(const float* __restrict__ src, float* __restrict__ dst,
                          int rows, int scols, int dcols) {
  int i = blockIdx.x * blockDim.x + threadIdx.x;
  if (i >= rows * dcols) return;
  int r = i / dcols, c = i - r * dcols;
  dst[i] = (c < scols) ? src[(size_t)r * scols + c] : 0.f;
}
// dst rows [0:drows) = src rows [0:srows) padded with zero rows
__global__ void k_padrows(const float* __restrict__ src, float* __restrict__ dst,
                          int srows, int drows, int cols) {
  int i = blockIdx.x * blockDim.x + threadIdx.x;
  if (i >= drows * cols) return;
  int r = i / cols;
  dst[i] = (r < srows) ? src[i] : 0.f;
}
__global__ void k_fill0(float* __restrict__ p, int n) {
  int i = blockIdx.x * blockDim.x + threadIdx.x;
  if (i < n) p[i] = 0.f;
}

// ---------------------------------------------------------------------------
// Elementwise / reduction kernels
// ---------------------------------------------------------------------------
__device__ __forceinline__ float sigmoidf_(float x) {
  return 1.0f / (1.0f + expf(-x));
}

__global__ void k_hprev(const float* __restrict__ hiddens, float* __restrict__ hprev) {
  int i = blockIdx.x * blockDim.x + threadIdx.x;
  if (i >= BT * HH) return;
  int bt = i / HH, j = i - bt * HH;
  int t = bt & (TT - 1);
  hprev[i] = (t == 0) ? 0.0f : hiddens[(size_t)(bt - 1) * HH + j];
}

__global__ void k_s(float* __restrict__ s, const float* __restrict__ cells) {
  int i = blockIdx.x * blockDim.x + threadIdx.x;
  if (i >= BT * HH) return;
  s[i] = sigmoidf_(s[i]) * tanhf(cells[i]);
}

__global__ void k_featmean(const float* __restrict__ V, float* __restrict__ fm) {
  int i = blockIdx.x * blockDim.x + threadIdx.x;
  if (i >= BB * CDIM) return;
  int b = i / CDIM, c = i - b * CDIM;
  float s = 0.0f;
  for (int k = 0; k < NPIX; ++k)
    s += V[((size_t)b * NPIX + k) * CDIM + c];
  fm[i] = s * (1.0f / (float)NPIX);
}

// cxt[bt,c] = sum_j tanh(fm[b,c]*Wfeat[j] + hWg2[bt,j]) * Wcxt[j]
__global__ void k_cxt(const float* __restrict__ fm, const float* __restrict__ hWg2,
                      const float* __restrict__ Wfeat, const float* __restrict__ Wcxt,
                      float* __restrict__ cxt) {
  __shared__ float sh_h[HH], sh_wf[HH], sh_wc[HH];
  int bt = blockIdx.x;
  int b  = bt >> 3;
  for (int j = threadIdx.x; j < HH; j += blockDim.x) {
    sh_h[j]  = hWg2[(size_t)bt * HH + j];
    sh_wf[j] = Wfeat[j];
    sh_wc[j] = Wcxt[j];
  }
  __syncthreads();
  for (int c = threadIdx.x; c < CDIM; c += blockDim.x) {
    float f = fm[(size_t)b * CDIM + c];
    float acc = 0.0f;
#pragma unroll 4
    for (int j = 0; j < HH; ++j)
      acc += tanhf(fmaf(f, sh_wf[j], sh_h[j])) * sh_wc[j];
    cxt[(size_t)bt * CDIM + c] = acc;
  }
}

__global__ void k_softmax_row(float* __restrict__ x, int ncols) {
  __shared__ float red[256];
  float* p = x + (size_t)blockIdx.x * ncols;
  float mx = -1e30f;
  for (int c = threadIdx.x; c < ncols; c += 256) mx = fmaxf(mx, p[c]);
  red[threadIdx.x] = mx; __syncthreads();
  for (int s = 128; s > 0; s >>= 1) {
    if (threadIdx.x < s) red[threadIdx.x] = fmaxf(red[threadIdx.x], red[threadIdx.x + s]);
    __syncthreads();
  }
  mx = red[0]; __syncthreads();
  float sum = 0.0f;
  for (int c = threadIdx.x; c < ncols; c += 256) {
    float e = expf(p[c] - mx); p[c] = e; sum += e;
  }
  red[threadIdx.x] = sum; __syncthreads();
  for (int s = 128; s > 0; s >>= 1) {
    if (threadIdx.x < s) red[threadIdx.x] += red[threadIdx.x + s];
    __syncthreads();
  }
  float inv = 1.0f / red[0];
  for (int c = threadIdx.x; c < ncols; c += 256) p[c] *= inv;
}

// wf[bt,k,c] = alpha0[bt,c] * V[b,k,c]
__global__ void k_wf(const float* __restrict__ alpha0, const float* __restrict__ V,
                     float* __restrict__ wf) {
  size_t i = (size_t)blockIdx.x * blockDim.x + threadIdx.x;
  if (i >= (size_t)BT * NPIX * CDIM) return;
  int c  = (int)(i & (CDIM - 1));
  size_t rk = i >> 11;               // bt*NPIX + k
  int bt = (int)(rk / NPIX);
  int k  = (int)(rk - (size_t)bt * NPIX);
  int b  = bt >> 3;
  wf[i] = alpha0[(size_t)bt * CDIM + c] * V[((size_t)b * NPIX + k) * CDIM + c];
}

// c_chan[bt*NPAD + k] = mean over CDIM of wf row (bt*NPIX + k)
__global__ void k_mean2048(const float* __restrict__ wf, float* __restrict__ out) {
  __shared__ float red[256];
  int row = blockIdx.x;              // 0..BT*NPIX-1
  const float* p = wf + (size_t)row * CDIM;
  float s = 0.0f;
  for (int c = threadIdx.x; c < CDIM; c += 256) s += p[c];
  red[threadIdx.x] = s; __syncthreads();
  for (int st = 128; st > 0; st >>= 1) {
    if (threadIdx.x < st) red[threadIdx.x] += red[threadIdx.x + st];
    __syncthreads();
  }
  if (threadIdx.x == 0) {
    int bt = row / NPIX, k = row - bt * NPIX;
    out[(size_t)bt * NPAD + k] = red[0] * (1.0f / (float)CDIM);
  }
}

// z_t[bt,k] = sum_j tanh(tanh(cont49[bt,k,j] + hWg[bt,j])) * Wh[j]   (ld NPAD)
__global__ void k_zt(const float* __restrict__ cont49, const float* __restrict__ hWg,
                     const float* __restrict__ Wh, float* __restrict__ z_t) {
  int i = blockIdx.x * blockDim.x + threadIdx.x;
  if (i >= BT * NPIX) return;
  int bt = i / NPIX, k = i - bt * NPIX;
  const float* row = cont49 + ((size_t)bt * NPIX + k) * NPAD;
  const float* g   = hWg + (size_t)bt * NPAD;
  float acc = 0.0f;
  for (int j = 0; j < NPIX; ++j)
    acc += tanhf(tanhf(row[j] + g[j])) * Wh[j];
  z_t[i] = acc;
}

__global__ void k_softmax49(const float* __restrict__ z_t, float* __restrict__ a_ws,
                            float* __restrict__ a_out) {
  __shared__ float sh[64];
  int bt = blockIdx.x;
  int k  = threadIdx.x;  // 0..63
  float v = (k < NPIX) ? z_t[(size_t)bt * NPIX + k] : -1e30f;
  sh[k] = v; __syncthreads();
  for (int s = 32; s > 0; s >>= 1) {
    if (k < s) sh[k] = fmaxf(sh[k], sh[k + s]);
    __syncthreads();
  }
  float mx = sh[0]; __syncthreads();
  float e = (k < NPIX) ? expf(v - mx) : 0.0f;
  sh[k] = e; __syncthreads();
  for (int s = 32; s > 0; s >>= 1) {
    if (k < s) sh[k] += sh[k + s];
    __syncthreads();
  }
  float a = e / sh[0];
  if (k < NPIX) {
    a_ws[(size_t)bt * NPIX + k]  = a;
    a_out[(size_t)bt * NPIX + k] = a;
  }
}

__global__ void k_cspatial(const float* __restrict__ alpha, const float* __restrict__ V,
                           float* __restrict__ c_spat) {
  int i = blockIdx.x * blockDim.x + threadIdx.x;
  if (i >= BT * CDIM) return;
  int bt = i >> 11;
  int d  = i & (CDIM - 1);
  int b  = bt >> 3;
  float acc = 0.0f;
  for (int k = 0; k < NPIX; ++k)
    acc += alpha[(size_t)bt * NPIX + k] * V[((size_t)b * NPIX + k) * CDIM + d];
  c_spat[i] = acc;
}

// beta[bt] = softmax(concat(z_t[bt,:], z_ext[bt]))[-1]   (contS/hWg ld NPAD)
__global__ void k_beta(const float* __restrict__ contS, const float* __restrict__ hWg,
                       const float* __restrict__ Wh, const float* __restrict__ z_t,
                       float* __restrict__ beta_out) {
  int bt = blockIdx.x * blockDim.x + threadIdx.x;
  if (bt >= BT) return;
  float ze = 0.0f;
  for (int j = 0; j < NPIX; ++j)
    ze += tanhf(contS[(size_t)bt * NPAD + j] + hWg[(size_t)bt * NPAD + j]) * Wh[j];
  float mx = ze;
  for (int k = 0; k < NPIX; ++k) mx = fmaxf(mx, z_t[(size_t)bt * NPIX + k]);
  float den = expf(ze - mx);
  for (int k = 0; k < NPIX; ++k) den += expf(z_t[(size_t)bt * NPIX + k] - mx);
  beta_out[bt] = expf(ze - mx) / den;
}

__global__ void k_chat(const float* __restrict__ sg, const float* __restrict__ cg,
                       const float* __restrict__ spat, const float* __restrict__ chan,
                       const float* __restrict__ hiddens, float* __restrict__ chplus) {
  int i = blockIdx.x * blockDim.x + threadIdx.x;
  if (i >= BT * HH) return;
  chplus[i] = sigmoidf_(sg[i]) * spat[i] + sigmoidf_(cg[i]) * chan[i] + hiddens[i];
}

// ---------------------------------------------------------------------------
// Launcher
// ---------------------------------------------------------------------------
extern "C" void kernel_launch(void* const* d_in, const int* in_sizes, int n_in,
                              void* d_out, int out_size, void* d_ws, size_t ws_size,
                              hipStream_t stream) {
  const float* x       = (const float*)d_in[0];
  const float* hiddens = (const float*)d_in[1];
  const float* cells   = (const float*)d_in[2];
  const float* V       = (const float*)d_in[3];
  const float* Wsx     = (const float*)d_in[4];
  const float* Wsh     = (const float*)d_in[5];
  const float* Wv      = (const float*)d_in[6];
  const float* Wg      = (const float*)d_in[7];
  const float* Ws      = (const float*)d_in[8];
  const float* Wh      = (const float*)d_in[9];
  const float* Wfeat   = (const float*)d_in[10];
  const float* Wcxt    = (const float*)d_in[11];
  const float* Wg2     = (const float*)d_in[12];
  const float* Wspat   = (const float*)d_in[13];
  const float* Wchan   = (const float*)d_in[14];
  const float* Wgvs    = (const float*)d_in[15];
  const float* Wgvc    = (const float*)d_in[16];
  const float* Wghs    = (const float*)d_in[17];
  const float* Wghc    = (const float*)d_in[18];
  const float* Wmlp    = (const float*)d_in[19];
  const float* bmlp    = (const float*)d_in[20];

  float* out       = (float*)d_out;
  float* out_alpha = out + (size_t)BT * VOCAB;
  float* out_beta  = out_alpha + (size_t)BT * NPIX;

  // workspace carve-up (fp32); sizes even up to alpha_t (odd), so later buffers only 4-byte aligned
  float* W = (float*)d_ws;
  size_t o = 0;
  float* s_buf    = W + o; o += (size_t)BT * HH;
  float* hprev    = W + o; o += (size_t)BT * HH;
  float* hWg2     = W + o; o += (size_t)BT * HH;
  float* hWg      = W + o; o += (size_t)BT * NPAD;
  float* fm       = W + o; o += (size_t)BB * CDIM;
  float* cxt      = W + o; o += (size_t)BT * CDIM;          // alpha0 in place
  float* wf       = W + o; o += (size_t)BT * NPIX * CDIM;
  float* cont49   = W + o; o += (size_t)BT * NPIX * NPAD;
  float* z_t      = W + o; o += (size_t)BT * NPIX;
  float* c_chan   = W + o; o += (size_t)BT * NPAD;
  float* c_spat   = W + o; o += (size_t)BT * CDIM;
  float* spat_i   = W + o; o += (size_t)BT * HH;
  float* chan_i   = W + o; o += (size_t)BT * HH;
  float* sg_pre   = W + o; o += (size_t)BT * HH;
  float* cg_pre   = W + o; o += (size_t)BT * HH;
  float* contS    = W + o; o += (size_t)BT * NPAD;
  float* chplus   = W + o; o += (size_t)BT * HH;
  float* alpha_t  = W + o; o += (size_t)BT * NPIX + 1;      // odd: later buffers lose 8-byte alignment
  float* Wv_pad   = W + o; o += (size_t)CDIM * NPAD;
  float* Wg_pad   = W + o; o += (size_t)HH * NPAD;
  float* Ws_pad   = W + o; o += (size_t)HH * NPAD;
  float* Wchan_p  = W + o; o += (size_t)NPAD * HH;

  const dim3 blk128(128), blk256(256), blk64(64);

  // 0) padded operand prep (independent)
  k_padcols<<<(CDIM * NPAD + 255) / 256, blk256, 0, stream>>>(Wv, Wv_pad, CDIM, NPIX, NPAD);
  k_padcols<<<(HH * NPAD + 255) / 256, blk256, 0, stream>>>(Wg, Wg_pad, HH, NPIX, NPAD);
  k_padcols<<<(HH * NPAD + 255) / 256, blk256, 0, stream>>>(Ws, Ws_pad, HH, NPIX, NPAD);
  k_padrows<<<(NPAD * HH + 255) / 256, blk256, 0, stream>>>(Wchan, Wchan_p, NPIX, NPAD, HH);
  k_fill0<<<(BT * NPAD + 255) / 256, blk256, 0, stream>>>(c_chan, BT * NPAD);

  // 1) h_prev
  k_hprev<<<(BT * HH + 255) / 256, blk256, 0, stream>>>(hiddens, hprev);

  // 2) s_pre = x@Wsx + hprev@Wsh
  gemm_wmma_bf16<<<dim3(8, 1), blk128, 0, stream>>>(x, Wsx, s_buf,
      HH, E2, E2, HH, HH, 0, nullptr);
  gemm_wmma_bf16<<<dim3(8, 1), blk128, 0, stream>>>(hprev, Wsh, s_buf,
      HH, HH, HH, HH, HH, 1, nullptr);

  // 3) s = sigmoid(s_pre) * tanh(cells)
  k_s<<<(BT * HH + 255) / 256, blk256, 0, stream>>>(s_buf, cells);

  // 4) hWg2 = h @ Wg2 ; hWg = h @ Wg_pad
  gemm_wmma_bf16<<<dim3(8, 1), blk128, 0, stream>>>(hiddens, Wg2, hWg2,
      HH, HH, HH, HH, HH, 0, nullptr);
  gemm_wmma_bf16<<<dim3(1, 1), blk128, 0, stream>>>(hiddens, Wg_pad, hWg,
      NPAD, HH, HH, NPAD, NPAD, 0, nullptr);

  // 5) featmean, cxt, alpha0
  k_featmean<<<(BB * CDIM + 255) / 256, blk256, 0, stream>>>(V, fm);
  k_cxt<<<BT, blk256, 0, stream>>>(fm, hWg2, Wfeat, Wcxt, cxt);
  k_softmax_row<<<BT, blk256, 0, stream>>>(cxt, CDIM);

  // 6) wf = alpha0 * V ; channel mean (into padded c_chan)
  {
    size_t n = (size_t)BT * NPIX * CDIM;
    k_wf<<<(unsigned)((n + 255) / 256), blk256, 0, stream>>>(cxt, V, wf);
  }
  k_mean2048<<<BT * NPIX, blk256, 0, stream>>>(wf, c_chan);

  // 7) cont49 = wf(3136x2048) @ Wv_pad(2048x64)
  gemm_wmma_bf16<<<dim3(1, (BT * NPIX) / 64), blk128, 0, stream>>>(wf, Wv_pad, cont49,
      NPAD, CDIM, CDIM, NPAD, NPAD, 0, nullptr);

  // 8) z_t, alpha_t (-> d_out), c_spatial
  k_zt<<<(BT * NPIX + 127) / 128, blk128, 0, stream>>>(cont49, hWg, Wh, z_t);
  k_softmax49<<<BT, blk64, 0, stream>>>(z_t, alpha_t, out_alpha);
  k_cspatial<<<(BT * CDIM + 255) / 256, blk256, 0, stream>>>(alpha_t, V, c_spat);

  // 9) spatial/channel info
  gemm_wmma_bf16<<<dim3(8, 1), blk128, 0, stream>>>(c_spat, Wspat, spat_i,
      HH, CDIM, CDIM, HH, HH, 0, nullptr);
  gemm_wmma_bf16<<<dim3(8, 1), blk128, 0, stream>>>(c_chan, Wchan_p, chan_i,
      HH, NPAD, NPAD, HH, HH, 0, nullptr);

  // 10) gates
  gemm_wmma_bf16<<<dim3(8, 1), blk128, 0, stream>>>(s_buf, Wgvs, sg_pre,
      HH, HH, HH, HH, HH, 0, nullptr);
  gemm_wmma_bf16<<<dim3(8, 1), blk128, 0, stream>>>(hiddens, Wghs, sg_pre,
      HH, HH, HH, HH, HH, 1, nullptr);
  gemm_wmma_bf16<<<dim3(8, 1), blk128, 0, stream>>>(s_buf, Wgvc, cg_pre,
      HH, HH, HH, HH, HH, 0, nullptr);
  gemm_wmma_bf16<<<dim3(8, 1), blk128, 0, stream>>>(hiddens, Wghc, cg_pre,
      HH, HH, HH, HH, HH, 1, nullptr);

  // 11) contS = s @ Ws_pad ; beta -> d_out
  gemm_wmma_bf16<<<dim3(1, 1), blk128, 0, stream>>>(s_buf, Ws_pad, contS,
      NPAD, HH, HH, NPAD, NPAD, 0, nullptr);
  k_beta<<<2, dim3(32), 0, stream>>>(contS, hWg, Wh, z_t, out_beta);

  // 12) c_hat + h
  k_chat<<<(BT * HH + 255) / 256, blk256, 0, stream>>>(sg_pre, cg_pre, spat_i,
                                                       chan_i, hiddens, chplus);

  // 13) scores = chplus @ Wmlp + bmlp -> d_out (N=10000, last block ragged by tiles)
  gemm_wmma_bf16<<<dim3((VOCAB + 63) / 64, 1), blk128, 0, stream>>>(chplus, Wmlp, out,
      VOCAB, HH, HH, VOCAB, VOCAB, 0, bmlp);

  (void)in_sizes; (void)n_in; (void)out_size; (void)ws_size;
}
